// CausalSelfAttentionWithRPE_86638080295185
// MI455X (gfx1250) — matrix operand order repaired and checked by reading, not compiled, not run on hardware
//
#include <hip/hip_runtime.h>
#include <hip/hip_bf16.h>
#include <math.h>

// ---------------------------------------------------------------------------
// CausalSelfAttentionWithRPE for MI455X (gfx1250, wave32, WMMA bf16 + TDM)
// B=4, T=1024, C=1024, H=16, d=64
// ---------------------------------------------------------------------------

typedef __bf16 bf16;
typedef __attribute__((ext_vector_type(16))) __bf16 v16bf;
typedef __attribute__((ext_vector_type(8)))  float  v8f;
typedef __attribute__((ext_vector_type(4)))  unsigned int v4u;
typedef __attribute__((ext_vector_type(8)))  int    v8i;
typedef __attribute__((ext_vector_type(4)))  int    v4i;

#define DEV __device__ __forceinline__

#if defined(__HIP_DEVICE_COMPILE__) && __has_builtin(__builtin_amdgcn_tensor_load_to_lds)
#define HAVE_TDM 1
#else
#define HAVE_TDM 0
#endif

DEV v8f vzero8() {
  v8f z;
#pragma unroll
  for (int e = 0; e < 8; ++e) z[e] = 0.0f;
  return z;
}

DEV v8f wmma_bf16f32(v16bf a, v16bf b, v8f c) {
  // D = A(16x32 bf16) * B(32x16 bf16) + C(16x16 f32)
  return __builtin_amdgcn_wmma_f32_16x16x32_bf16(
      false, a, false, b, (short)0, c, false, false);
}

// A operand: row-major [M,K]; lane l<16 -> row=l, K = {k0+0..7, k0+16..23}
//            lane l>=16 -> row=l-16, K = {k0+8..15, k0+24..31}
DEV v16bf load_frag_a(const bf16* base, int ld, int row0, int k0) {
  int lane = threadIdx.x & 31;
  int hf = lane >> 4;
  const bf16* p = base + (size_t)(row0 + (lane & 15)) * ld + k0 + hf * 8;
  union { float4 f[2]; v16bf v; } u;
  u.f[0] = *(const float4*)(p);
  u.f[1] = *(const float4*)(p + 16);
  return u.v;
}

// B operand: [N,K] row-major (i.e. B^T of the math B[K,N]);
// lanes 0-15 hold K={k0+0..7, k0+16..23} of column n=lane; lanes 16-31 hold K={k0+8..15, k0+24..31}
DEV v16bf load_frag_b(const bf16* base, int ld, int n0, int k0) {
  int lane = threadIdx.x & 31;
  int hf = lane >> 4;
  const bf16* p = base + (size_t)(n0 + (lane & 15)) * ld + k0 + hf * 8;
  union { float4 f[2]; v16bf v; } u;
  u.f[0] = *(const float4*)(p);
  u.f[1] = *(const float4*)(p + 16);
  return u.v;
}

#if HAVE_TDM
// LDS byte offset of a __shared__ object (generic -> AS(3) -> int)
DEV unsigned int lds_offset(const void* p) {
  return (unsigned int)(unsigned long long)(const __attribute__((address_space(3))) char*)p;
}

// 2-D TDM load: tile_d0 contiguous elements (bf16) per row, tile_d1 rows,
// row stride `stride_elems`, packed into LDS at lds_off.
// D# layout per CDNA5 ISA.
DEV void tdm_load_2d(unsigned int lds_off, const void* gaddr,
                     unsigned int tile_d0, unsigned int tile_d1,
                     unsigned int stride_elems) {
  unsigned long long ga = (unsigned long long)gaddr;
  v4u g0;
  g0[0] = 1u;                                          // count=1, user descriptor
  g0[1] = lds_off;                                     // lds_addr (bytes)
  g0[2] = (unsigned int)(ga & 0xffffffffull);          // global_addr[31:0]
  g0[3] = (unsigned int)((ga >> 32) & 0x1ffffffull)    // global_addr[56:32]
          | (2u << 30);                                // type = 2 ("image")
  v8i g1;
  g1[0] = (int)(1u << 16);                             // data_size=1 -> 2 bytes
  g1[1] = (int)((tile_d0 & 0xffffu) << 16);            // tensor_dim0[15:0]
  g1[2] = (int)(((tile_d0 >> 16) & 0xffffu) |          // tensor_dim0[31:16]
                ((tile_d1 & 0xffffu) << 16));          // tensor_dim1[15:0]
  g1[3] = (int)(((tile_d1 >> 16) & 0xffffu) |          // tensor_dim1[31:16]
                ((tile_d0 & 0xffffu) << 16));          // tile_dim0
  g1[4] = (int)(tile_d1 & 0xffffu);                    // tile_dim1 (tile_dim2=0)
  g1[5] = (int)stride_elems;                           // tensor_dim0_stride[31:0]
  g1[6] = 0;                                           // stride[47:32] | dim1_stride lo
  g1[7] = 0;
  v4i z4 = {0, 0, 0, 0};
#if __clang_major__ >= 23
  v8i z8 = {0, 0, 0, 0, 0, 0, 0, 0};
  __builtin_amdgcn_tensor_load_to_lds(g0, g1, z4, z4, z8, 0);
#else
  __builtin_amdgcn_tensor_load_to_lds(g0, g1, z4, z4, 0);
#endif
}

DEV void tdm_wait() { __builtin_amdgcn_s_wait_tensorcnt(0); }
#endif  // HAVE_TDM

// ---------------------------------------------------------------------------
// Conversions / layout prep
// ---------------------------------------------------------------------------
__global__ void cvt_f32_bf16(const float* __restrict__ in, bf16* __restrict__ out, int n) {
  int i = blockIdx.x * 256 + threadIdx.x;
  if (i < n) out[i] = (bf16)in[i];
}

// out[c][r] = in[r][c]   (in: R x Cc)
__global__ void cvt_transpose_bf16(const float* __restrict__ in, bf16* __restrict__ out,
                                   int R, int Cc) {
  int i = blockIdx.x * 256 + threadIdx.x;
  if (i < R * Cc) {
    int r = i / Cc, c = i - r * Cc;
    out[(size_t)c * R + r] = (bf16)in[i];
  }
}

// ekp[1088][64]: row = 32 + rel, zero padding outside rel in [0,1024)
__global__ void build_ekp(const float* __restrict__ embk, bf16* __restrict__ ekp) {
  int i = blockIdx.x * 256 + threadIdx.x;
  if (i < 1088 * 64) {
    int r = i >> 6, d = i & 63;
    int rel = r - 32;
    ekp[i] = (rel >= 0 && rel < 1024) ? (bf16)embk[rel * 64 + d] : (bf16)0.0f;
  }
}

// evT[64][1056]: col = 31 + rel (transposed embv), zero padded
__global__ void build_evt(const float* __restrict__ embv, bf16* __restrict__ evT) {
  int i = blockIdx.x * 256 + threadIdx.x;
  if (i < 64 * 1056) {
    int d = i / 1056, c = i - d * 1056;
    int rel = c - 31;
    evT[i] = (rel >= 0 && rel < 1024) ? (bf16)embv[rel * 64 + d] : (bf16)0.0f;
  }
}

// ---------------------------------------------------------------------------
// Shared GEMM block body: block = 4 waves, tile M=64 (16/wave), N=64, K=1024.
// B strip (64 N-rows x 32 K) is staged into double-buffered LDS: TDM by wave 0
// when available, else cooperative direct copy.
// ---------------------------------------------------------------------------
template <typename EPI>
DEV void gemm_block_64x64(const bf16* __restrict__ A, const bf16* __restrict__ Bm,
                          int ldb, EPI epilogue) {
  __shared__ __align__(16) bf16 Bs[2][64 * 32];
  int wave = threadIdx.x >> 5;
  int m0 = blockIdx.x * 64 + wave * 16;
  int n0 = blockIdx.y * 64;

  v8f acc[4];
#pragma unroll
  for (int i = 0; i < 4; ++i) acc[i] = vzero8();

#if HAVE_TDM
  if (threadIdx.x < 32) {
    tdm_load_2d(lds_offset(&Bs[0][0]), Bm + (size_t)n0 * ldb, 32, 64, ldb);
    tdm_wait();
  }
  __syncthreads();
  for (int k0 = 0; k0 < 1024; k0 += 32) {
    int cur = (k0 >> 5) & 1;
    if (threadIdx.x < 32 && k0 + 32 < 1024)
      tdm_load_2d(lds_offset(&Bs[cur ^ 1][0]),
                  Bm + (size_t)n0 * ldb + (k0 + 32), 32, 64, ldb);
    v16bf a = load_frag_a(A, 1024, m0, k0);
#pragma unroll
    for (int nt = 0; nt < 4; ++nt)
      acc[nt] = wmma_bf16f32(a, load_frag_b(&Bs[cur][0], 32, nt * 16, 0), acc[nt]);
    if (threadIdx.x < 32) tdm_wait();   // next buffer ready before swap
    __syncthreads();                    // all waves done reading cur
  }
#else
  // Fallback: cooperative copy of the B strip, single-buffered.
  for (int k0 = 0; k0 < 1024; k0 += 32) {
    __syncthreads();
    {
      int r = threadIdx.x >> 1, half = threadIdx.x & 1;   // 128 threads: 64 rows x 2
      const float4* src = (const float4*)(Bm + (size_t)(n0 + r) * ldb + k0 + half * 16);
      float4* dst = (float4*)(&Bs[0][r * 32 + half * 16]);
      dst[0] = src[0];
      dst[1] = src[1];
    }
    __syncthreads();
    v16bf a = load_frag_a(A, 1024, m0, k0);
#pragma unroll
    for (int nt = 0; nt < 4; ++nt)
      acc[nt] = wmma_bf16f32(a, load_frag_b(&Bs[0][0], 32, nt * 16, 0), acc[nt]);
  }
#endif
  epilogue(acc, m0, n0);
}

// ---------------------------------------------------------------------------
// QKV GEMM: [4096,1024] x [1024,3072] -> scatter q/k/v into per-head layouts
//   qb,kb: [B,H,T,64] bf16 (k pre-scaled by 1/8);  vT: [B,H,64,T] bf16
// ---------------------------------------------------------------------------
__global__ __launch_bounds__(128) void qkv_gemm(
    const bf16* __restrict__ xb, const bf16* __restrict__ waT,
    bf16* __restrict__ qb, bf16* __restrict__ kb, bf16* __restrict__ vT) {
  int lane = threadIdx.x & 31;
  int hf = lane >> 4, ln = lane & 15;
  gemm_block_64x64(xb, waT, 1024, [&](v8f* acc, int m0, int n0) {
#pragma unroll
    for (int nt = 0; nt < 4; ++nt) {
      int n = n0 + nt * 16 + ln;
      int sec = n >> 10;  // 0=q, 1=k, 2=v
      int cn = n & 1023;
      int h = cn >> 6, dd = cn & 63;
#pragma unroll
      for (int v = 0; v < 8; ++v) {
        int m = m0 + v + 8 * hf;
        int b = m >> 10, t = m & 1023;
        float val = acc[nt][v];
        size_t hb = (size_t)(b * 16 + h);
        if (sec == 0)      qb[(hb * 1024 + t) * 64 + dd] = (bf16)val;
        else if (sec == 1) kb[(hb * 1024 + t) * 64 + dd] = (bf16)(val * 0.125f);
        else               vT[(hb * 64 + dd) * 1024 + t] = (bf16)val;
      }
    }
  });
}

// ---------------------------------------------------------------------------
// Output projection: out[4096,1024] f32 = yb @ w_proj (via wpT [N,K])
// ---------------------------------------------------------------------------
__global__ __launch_bounds__(128) void proj_gemm(
    const bf16* __restrict__ yb, const bf16* __restrict__ wpT,
    float* __restrict__ out) {
  int lane = threadIdx.x & 31;
  int hf = lane >> 4, ln = lane & 15;
  gemm_block_64x64(yb, wpT, 1024, [&](v8f* acc, int m0, int n0) {
#pragma unroll
    for (int nt = 0; nt < 4; ++nt) {
      int n = n0 + nt * 16 + ln;
#pragma unroll
      for (int v = 0; v < 8; ++v) {
        int m = m0 + v + 8 * hf;
        out[(size_t)m * 1024 + n] = acc[nt][v];
      }
    }
  });
}

// ---------------------------------------------------------------------------
// Flash attention with relative-position terms.
// 1 wave per block; block = (b,h, q-tile of 16 rows); stream s in chunks of 32.
// ---------------------------------------------------------------------------
__global__ __launch_bounds__(32) void attn_kernel(
    const bf16* __restrict__ qb, const bf16* __restrict__ kb,
    const bf16* __restrict__ vT, const bf16* __restrict__ ekp,
    const bf16* __restrict__ evT, bf16* __restrict__ yb) {
  __shared__ __align__(16) float Pbuf[16 * 32];
  __shared__ __align__(16) bf16  pAbuf[16 * 32];
  __shared__ __align__(16) bf16  pskew[16 * 32];

  int lane = threadIdx.x & 31;
  int hf = lane >> 4, ln = lane & 15;
  int bh = blockIdx.x >> 6;
  int t0 = (blockIdx.x & 63) << 4;
  int b = bh >> 4, h = bh & 15;

  const bf16* qbase = qb + (size_t)bh * 1024 * 64;
  const bf16* kbase = kb + (size_t)bh * 1024 * 64;
  const bf16* vbase = vT + (size_t)bh * 64 * 1024;

  v16bf qa0 = load_frag_a(qbase, 64, t0, 0);
  v16bf qa1 = load_frag_a(qbase, 64, t0, 32);

  float mrow[8], lrow[8];
  v8f o[4];
#pragma unroll
  for (int v = 0; v < 8; ++v) { mrow[v] = -1e30f; lrow[v] = 0.0f; }
#pragma unroll
  for (int nt = 0; nt < 4; ++nt) o[nt] = vzero8();

  for (int s0 = 0; s0 < t0 + 16; s0 += 32) {
    float sc[2][8];
#pragma unroll
    for (int j = 0; j < 2; ++j) {
      int scol = s0 + 16 * j;
      if (scol <= t0) {
        // S1 = (q * k^T) * scale   (k pre-scaled by 0.125)
        v8f c = vzero8();
        c = wmma_bf16f32(qa0, load_frag_b(kbase, 64, scol, 0), c);
        c = wmma_bf16f32(qa1, load_frag_b(kbase, 64, scol, 32), c);
        // P[t', r'] = q[t0+t'] . embk[(t0-scol-15) + r'],  r' in [0,31]
        int rb = t0 - scol + 17;  // = 32 + (t0-scol-15), padded row base
        v8f p0 = vzero8(), p1 = vzero8();
        p0 = wmma_bf16f32(qa0, load_frag_b(ekp, 64, rb, 0), p0);
        p0 = wmma_bf16f32(qa1, load_frag_b(ekp, 64, rb, 32), p0);
        p1 = wmma_bf16f32(qa0, load_frag_b(ekp, 64, rb + 16, 0), p1);
        p1 = wmma_bf16f32(qa1, load_frag_b(ekp, 64, rb + 16, 32), p1);
        __syncthreads();
#pragma unroll
        for (int v = 0; v < 8; ++v) {
          int tp = v + 8 * hf;
          Pbuf[tp * 32 + ln] = p0[v];
          Pbuf[tp * 32 + 16 + ln] = p1[v];
        }
        __syncthreads();
#pragma unroll
        for (int v = 0; v < 8; ++v) {
          int tp = v + 8 * hf;
          float s = c[v] + Pbuf[tp * 32 + 15 + tp - ln];  // skew gather
          sc[j][v] = ((scol + ln) <= (t0 + tp)) ? s : -1e30f;
        }
        __syncthreads();
      } else {
#pragma unroll
        for (int v = 0; v < 8; ++v) sc[j][v] = -1e30f;
      }
    }

    // ---- online softmax (rows spread over 16-lane halves in C layout) ----
    float rm[8], alpha[8], pc0[8], pc1[8], rs[8];
#pragma unroll
    for (int v = 0; v < 8; ++v) rm[v] = fmaxf(sc[0][v], sc[1][v]);
#pragma unroll
    for (int off = 1; off <= 8; off <<= 1)
#pragma unroll
      for (int v = 0; v < 8; ++v) rm[v] = fmaxf(rm[v], __shfl_xor(rm[v], off, 32));
#pragma unroll
    for (int v = 0; v < 8; ++v) {
      float nm = fmaxf(mrow[v], rm[v]);
      alpha[v] = __expf(mrow[v] - nm);
      mrow[v] = nm;
      pc0[v] = __expf(sc[0][v] - nm);
      pc1[v] = __expf(sc[1][v] - nm);
      rs[v] = pc0[v] + pc1[v];
    }
#pragma unroll
    for (int off = 1; off <= 8; off <<= 1)
#pragma unroll
      for (int v = 0; v < 8; ++v) rs[v] += __shfl_xor(rs[v], off, 32);
#pragma unroll
    for (int v = 0; v < 8; ++v) lrow[v] = lrow[v] * alpha[v] + rs[v];
#pragma unroll
    for (int nt = 0; nt < 4; ++nt)
#pragma unroll
      for (int v = 0; v < 8; ++v) o[nt][v] *= alpha[v];

    // ---- O += P(16x32) @ V(32x64): route p through LDS into A layout ----
    __syncthreads();
#pragma unroll
    for (int v = 0; v < 8; ++v) {
      int tp = v + 8 * hf;
      pAbuf[tp * 32 + ln] = (bf16)pc0[v];
      pAbuf[tp * 32 + 16 + ln] = (bf16)pc1[v];
    }
    __syncthreads();
    {
      v16bf pa = load_frag_a((const bf16*)pAbuf, 32, 0, 0);
#pragma unroll
      for (int nt = 0; nt < 4; ++nt)
        o[nt] = wmma_bf16f32(pa, load_frag_b(vbase, 1024, nt * 16, s0), o[nt]);
    }

    // ---- O += skew(P) @ embv : per 16-col subtile ----
#pragma unroll
    for (int j = 0; j < 2; ++j) {
      int scol = s0 + 16 * j;
      if (scol <= t0) {
        __syncthreads();
#pragma unroll
        for (int i = 0; i < 16; ++i) pskew[i * 32 + lane] = (bf16)0.0f;
        __syncthreads();
#pragma unroll
        for (int v = 0; v < 8; ++v) {
          int tp = v + 8 * hf;
          pskew[tp * 32 + 15 + tp - ln] = (bf16)(j ? pc1[v] : pc0[v]);
        }
        __syncthreads();
        v16bf ps = load_frag_a((const bf16*)pskew, 32, 0, 0);
        int kev = t0 - scol + 16;  // = 31 + (t0-scol-15), padded col base
#pragma unroll
        for (int nt = 0; nt < 4; ++nt)
          o[nt] = wmma_bf16f32(ps, load_frag_b(evT, 1056, nt * 16, kev), o[nt]);
      }
    }
  }

  // ---- epilogue: y = O / l, write bf16 into [B*T, C] with head offset ----
#pragma unroll
  for (int nt = 0; nt < 4; ++nt)
#pragma unroll
    for (int v = 0; v < 8; ++v) {
      int tp = v + 8 * hf;
      int t = t0 + tp;
      float val = o[nt][v] / lrow[v];
      yb[((size_t)(b * 1024 + t)) * 1024 + h * 64 + nt * 16 + ln] = (bf16)val;
    }
}

// ---------------------------------------------------------------------------
// Workspace layout (bytes)
// ---------------------------------------------------------------------------
static constexpr size_t OFF_XB  = 0;                                   // 4096x1024 bf16
static constexpr size_t OFF_WAT = OFF_XB  + (size_t)4096 * 1024 * 2;   // 3072x1024
static constexpr size_t OFF_WPT = OFF_WAT + (size_t)3072 * 1024 * 2;   // 1024x1024
static constexpr size_t OFF_EKP = OFF_WPT + (size_t)1024 * 1024 * 2;   // 1088x64
static constexpr size_t OFF_EVT = OFF_EKP + (size_t)1088 * 64 * 2;     // 64x1056
static constexpr size_t OFF_QB  = OFF_EVT + (size_t)64 * 1056 * 2;     // B,H,T,64
static constexpr size_t OFF_KB  = OFF_QB  + (size_t)4 * 16 * 1024 * 64 * 2;
static constexpr size_t OFF_VT  = OFF_KB  + (size_t)4 * 16 * 1024 * 64 * 2;
static constexpr size_t OFF_YB  = OFF_VT  + (size_t)4 * 16 * 1024 * 64 * 2;

extern "C" void kernel_launch(void* const* d_in, const int* in_sizes, int n_in,
                              void* d_out, int out_size, void* d_ws, size_t ws_size,
                              hipStream_t stream) {
  const float* x      = (const float*)d_in[0];
  const float* w_attn = (const float*)d_in[1];
  const float* w_proj = (const float*)d_in[2];
  const float* embk   = (const float*)d_in[3];
  const float* embv   = (const float*)d_in[4];
  float* out = (float*)d_out;

  char* ws = (char*)d_ws;
  bf16* xb  = (bf16*)(ws + OFF_XB);
  bf16* waT = (bf16*)(ws + OFF_WAT);
  bf16* wpT = (bf16*)(ws + OFF_WPT);
  bf16* ekp = (bf16*)(ws + OFF_EKP);
  bf16* evT = (bf16*)(ws + OFF_EVT);
  bf16* qb  = (bf16*)(ws + OFF_QB);
  bf16* kb  = (bf16*)(ws + OFF_KB);
  bf16* vT  = (bf16*)(ws + OFF_VT);
  bf16* yb  = (bf16*)(ws + OFF_YB);

  // layout prep
  cvt_f32_bf16<<<(4096 * 1024 + 255) / 256, 256, 0, stream>>>(x, xb, 4096 * 1024);
  cvt_transpose_bf16<<<(1024 * 3072 + 255) / 256, 256, 0, stream>>>(w_attn, waT, 1024, 3072);
  cvt_transpose_bf16<<<(1024 * 1024 + 255) / 256, 256, 0, stream>>>(w_proj, wpT, 1024, 1024);
  build_ekp<<<(1088 * 64 + 255) / 256, 256, 0, stream>>>(embk, ekp);
  build_evt<<<(64 * 1056 + 255) / 256, 256, 0, stream>>>(embv, evT);

  // qkv projection + per-head scatter (k scaled by 1/sqrt(d))
  qkv_gemm<<<dim3(64, 48), 128, 0, stream>>>(xb, waT, qb, kb, vT);

  // flash attention with RPE: one wave per (b,h,q-tile)
  attn_kernel<<<4 * 16 * 64, 32, 0, stream>>>(qb, kb, vT, ekp, evT, yb);

  // output projection to f32
  proj_gemm<<<dim3(64, 16), 128, 0, stream>>>(yb, wpT, out);
}
